// RoPEAttention_87256555585647
// MI455X (gfx1250) — matrix operand order repaired and checked by reading, not compiled, not run on hardware
//
#include <hip/hip_runtime.h>
#include <hip/hip_bf16.h>

// Problem constants (from reference setup_inputs)
#define BB   2
#define SS   4096
#define HH   1024
#define NROW (BB*SS)          // 8192 total rows across batch

typedef __attribute__((ext_vector_type(16))) __bf16 v16bf;
typedef __attribute__((ext_vector_type(8)))  __bf16 v8bf;
typedef __attribute__((ext_vector_type(8)))  float  v8f;

union ABFrag { v16bf v; v8bf h[2]; };

// ---------------------------------------------------------------------------
// Stage 0: fp32 -> bf16 conversion (x and the three weight matrices)
// ---------------------------------------------------------------------------
__global__ void cvt_f32_bf16(const float* __restrict__ src,
                             __bf16* __restrict__ dst, int n) {
  int i = blockIdx.x * blockDim.x + threadIdx.x;
  int stride = gridDim.x * blockDim.x;
  for (; i < n; i += stride) dst[i] = (__bf16)src[i];
}

// ---------------------------------------------------------------------------
// Stage 1: fused QKV projection + bias + RoPE. One 16x64 output tile per wave
// (A-fragment reused across 4 WMMAs per K-step -> 2x arithmetic intensity).
//   grid = (NROW/16, HH/256, 3 {q,k,v}), block = 128 (4 waves)
//   q,k stored row-major bf16; v stored transposed (b, h, t) bf16 so that
//   the P*V B-fragments in stage 2 are contiguous 32B loads.
// CDNA5 WMMA bf16 A-layout (16x32): lane m(0..15)/m+16 holds row m,
//   halves 0..7  = K (kk + 16*(lane>>4)) + {0..7}  (contiguous k order used here,
//   halves 8..15 = K (kk + 16*(lane>>4)) + {8..15}  not AMD's table order)
// B as loaded here (32x16, same k order as A): lane l holds column l&15, K = kk + 16*(l>>4) + {0..15}
// C/D layout: vgpr r holds M = r + 8*(l>>4), N = l&15.
// ---------------------------------------------------------------------------
__global__ __launch_bounds__(128) void qkv_rope_kernel(
    const __bf16* __restrict__ xb,
    const __bf16* __restrict__ wqb, const __bf16* __restrict__ wkb,
    const __bf16* __restrict__ wvb,
    const float* __restrict__ bq, const float* __restrict__ bk,
    const float* __restrict__ bv,
    __bf16* __restrict__ qb, __bf16* __restrict__ kbuf,
    __bf16* __restrict__ vt)
{
  const int lane = threadIdx.x & 31;
  const int w    = threadIdx.x >> 5;
  const int h16  = lane >> 4;
  const int ln   = lane & 15;
  const int rowBase = blockIdx.x * 16;
  const int colBase = blockIdx.y * 256 + w * 64;
  const int which   = blockIdx.z;

  const __bf16* wb   = (which == 0) ? wqb : (which == 1) ? wkb : wvb;
  const float*  bias = (which == 0) ? bq  : (which == 1) ? bk  : bv;

  v8f zf = {};
  v8f acc[4];
#pragma unroll
  for (int t = 0; t < 4; ++t) acc[t] = zf;

  for (int kk = 0; kk < HH; kk += 32) {
    ABFrag a;
    const __bf16* xr = xb + (size_t)(rowBase + ln) * HH + kk + 16 * h16;
    a.h[0] = *(const v8bf*)(xr);
    a.h[1] = *(const v8bf*)(xr + 8);
#pragma unroll
    for (int t = 0; t < 4; ++t) {
      v16bf bfr = *(const v16bf*)(wb + (size_t)(colBase + 16 * t + ln) * HH +
                                  kk + 16 * h16);
      acc[t] = __builtin_amdgcn_wmma_f32_16x16x32_bf16(false, a.v, false, bfr,
                                                       (short)0, acc[t],
                                                       false, false);
    }
  }

#pragma unroll
  for (int t = 0; t < 4; ++t) {
    const int c    = colBase + 16 * t + ln;
    const float bval = bias[c];
#pragma unroll
    for (int r = 0; r < 8; ++r) acc[t][r] += bval;

    if (which < 2) {
      // RoPE: t[2i]==t[2i+1] = (i even? sin : cos)(pos * 10000^(-(i>>1)/256))
      const int   j    = c >> 1;
      const int   i2   = j >> 1;
      const float freq = exp2f(-(float)i2 * (13.287712379549449f / 256.f));
#pragma unroll
      for (int r = 0; r < 8; ++r) {
        float own     = acc[t][r];
        float partner = __shfl_xor(own, 1, 32);   // pair column c^1, same row
        float pos     = (float)((rowBase + r + 8 * h16) & (SS - 1));
        float rad     = pos * freq;
        float tv      = (j & 1) ? __cosf(rad) : __sinf(rad);
        acc[t][r] = tv * ((c & 1) ? (own + partner) : (own - partner));
      }
      __bf16* dst = (which == 0) ? qb : kbuf;
#pragma unroll
      for (int r = 0; r < 8; ++r) {
        int row = rowBase + r + 8 * h16;
        dst[(size_t)row * HH + c] = (__bf16)acc[t][r];
      }
    } else {
      const int bidx = rowBase >> 12;             // rowBase / SS
#pragma unroll
      for (int r = 0; r < 8; ++r) {
        int row = rowBase + r + 8 * h16;
        int tt  = row & (SS - 1);
        vt[((size_t)bidx * HH + c) * SS + tt] = (__bf16)acc[t][r];
      }
    }
  }
}

// ---------------------------------------------------------------------------
// Stage 2: streaming (flash) attention. 16 query rows per 256-thread block
// (8 waves). Wave w owns key sub-column [16w,16w+16) of each 128-key block
// for scores, and output-column slab [128w, 128w+128) for P*V.
// Q tile staged into LDS via CDNA5 async global->LDS loads (ASYNCcnt).
// ---------------------------------------------------------------------------
#define QPITCH (HH + 8)
#define PPITCH (128 + 8)

__global__ __launch_bounds__(256) void attn_kernel(
    const __bf16* __restrict__ qb, const __bf16* __restrict__ kb,
    const __bf16* __restrict__ vt, float* __restrict__ out)
{
  __shared__ __bf16 q_lds[16 * QPITCH];     // 16 x 1024 bf16, padded
  __shared__ __bf16 p_lds[16 * PPITCH];     // 16 x 128 bf16, padded
  __shared__ float  wavemax[8][16];
  __shared__ float  wavesum[8][16];
  __shared__ float  rowM[16], rowL[16], rowScale[16];

  const int tid  = threadIdx.x;
  const int lane = tid & 31;
  const int w    = tid >> 5;
  const int h16  = lane >> 4;
  const int ln   = lane & 15;
  const int qrowBase = blockIdx.x * 16;
  const int b        = qrowBase >> 12;
  const size_t kbase = (size_t)b * SS;

  // stage 16 x 1024 bf16 q tile into LDS with async global->LDS B128 copies;
  // per-lane LDS destination addresses preserve the padded pitch.
  for (int i = tid; i < 16 * (HH / 8); i += 256) {
    int row = i >> 7, cc = i & 127;
    unsigned dstOff = (unsigned)(size_t)(q_lds + row * QPITCH + cc * 8);
    const __bf16* src = qb + (size_t)(qrowBase + row) * HH + cc * 8;
    asm volatile("global_load_async_to_lds_b128 %0, %1, off"
                 :: "v"(dstOff), "v"(src)
                 : "memory");
  }
  asm volatile("s_wait_asynccnt 0x0" ::: "memory");
  if (tid < 16) { rowM[tid] = -1e30f; rowL[tid] = 0.f; }
  __syncthreads();

  v8f zf = {};
  v8f o[8];
#pragma unroll
  for (int f = 0; f < 8; ++f) o[f] = zf;

  const float rsc = 0.044194173824159216f;   // 1/sqrt(HEAD_SIZE=512)

  for (int kblk = 0; kblk < SS; kblk += 128) {
    // ---- scores: s(16x16) = q(16x1024) . k^T for this wave's 16 keys ----
    v8f s = zf;
    const __bf16* krow = kb + (kbase + kblk + 16 * w + ln) * HH;
    __builtin_prefetch(kb + (kbase + ((kblk + 128) & (SS - 1)) + 16 * w + ln) * HH, 0, 0);
    for (int kk = 0; kk < HH; kk += 32) {
      ABFrag a;
      const __bf16* qr = q_lds + ln * QPITCH + kk + 16 * h16;
      a.h[0] = *(const v8bf*)(qr);
      a.h[1] = *(const v8bf*)(qr + 8);
      v16bf bf = *(const v16bf*)(krow + kk + 16 * h16);
      s = __builtin_amdgcn_wmma_f32_16x16x32_bf16(false, a.v, false, bf,
                                                  (short)0, s, false, false);
    }
#pragma unroll
    for (int r = 0; r < 8; ++r) s[r] *= rsc;

    // ---- per-wave row max over this wave's 16 keys ----
#pragma unroll
    for (int r = 0; r < 8; ++r) {
      float v = s[r];
      v = fmaxf(v, __shfl_xor(v, 8, 32));
      v = fmaxf(v, __shfl_xor(v, 4, 32));
      v = fmaxf(v, __shfl_xor(v, 2, 32));
      v = fmaxf(v, __shfl_xor(v, 1, 32));
      if (ln == 0) wavemax[w][r + 8 * h16] = v;
    }
    __syncthreads();

    // ---- online-softmax running max / rescale factor (one thread per row)
    if (tid < 16) {
      float mb = wavemax[0][tid];
#pragma unroll
      for (int ww = 1; ww < 8; ++ww) mb = fmaxf(mb, wavemax[ww][tid]);
      float Mn = fmaxf(rowM[tid], mb);
      rowScale[tid] = __expf(rowM[tid] - Mn);
      rowM[tid]     = Mn;
    }
    __syncthreads();

    // ---- exp, accumulator rescale, partial sums, publish P (bf16) ----
#pragma unroll
    for (int r = 0; r < 8; ++r) {
      int   row = r + 8 * h16;
      float p   = __expf(s[r] - rowM[row]);
      float sc  = rowScale[row];
#pragma unroll
      for (int f = 0; f < 8; ++f) o[f][r] *= sc;
      float v = p;
      v += __shfl_xor(v, 8, 32);
      v += __shfl_xor(v, 4, 32);
      v += __shfl_xor(v, 2, 32);
      v += __shfl_xor(v, 1, 32);
      if (ln == 0) wavesum[w][row] = v;
      p_lds[row * PPITCH + 16 * w + ln] = (__bf16)p;
    }
    __syncthreads();

    if (tid < 16) {
      float tot = 0.f;
#pragma unroll
      for (int ww = 0; ww < 8; ++ww) tot += wavesum[ww][tid];
      rowL[tid] = rowL[tid] * rowScale[tid] + tot;
    }

    // ---- o(16x128 slab) += P(16x128) . V(128 x slab), V is transposed ----
    for (int kk2 = 0; kk2 < 128; kk2 += 32) {
      ABFrag a;
      const __bf16* pr = p_lds + ln * PPITCH + kk2 + 16 * h16;
      a.h[0] = *(const v8bf*)(pr);
      a.h[1] = *(const v8bf*)(pr + 8);
#pragma unroll
      for (int f = 0; f < 8; ++f) {
        int c0 = w * 128 + f * 16;
        v16bf bf = *(const v16bf*)(vt + ((size_t)b * HH + c0 + ln) * SS +
                                   kblk + kk2 + 16 * h16);
        o[f] = __builtin_amdgcn_wmma_f32_16x16x32_bf16(false, a.v, false, bf,
                                                       (short)0, o[f], false, false);
      }
    }
  }
  __syncthreads();   // make final rowL visible

  // ---- normalize and store fp32 output ----
#pragma unroll
  for (int f = 0; f < 8; ++f) {
#pragma unroll
    for (int r = 0; r < 8; ++r) {
      int row = r + 8 * h16;
      out[(size_t)(qrowBase + row) * HH + w * 128 + f * 16 + ln] =
          o[f][r] / rowL[row];
    }
  }
}

// ---------------------------------------------------------------------------
// Launch
// ---------------------------------------------------------------------------
extern "C" void kernel_launch(void* const* d_in, const int* in_sizes, int n_in,
                              void* d_out, int out_size, void* d_ws, size_t ws_size,
                              hipStream_t stream) {
  const float* x  = (const float*)d_in[0];
  const float* Wq = (const float*)d_in[1];
  const float* bq = (const float*)d_in[2];
  const float* Wk = (const float*)d_in[3];
  const float* bk = (const float*)d_in[4];
  const float* Wv = (const float*)d_in[5];
  const float* bv = (const float*)d_in[6];
  float* out = (float*)d_out;

  // Workspace layout (bf16 buffers), total ~70 MiB
  const size_t XB = (size_t)NROW * HH * sizeof(__bf16);   // 16 MiB
  const size_t WB = (size_t)HH * HH * sizeof(__bf16);     //  2 MiB
  char* ws = (char*)d_ws;
  __bf16* xb   = (__bf16*)(ws);
  __bf16* wqb  = (__bf16*)(ws + XB);
  __bf16* wkb  = (__bf16*)(ws + XB + WB);
  __bf16* wvb  = (__bf16*)(ws + XB + 2 * WB);
  __bf16* qb   = (__bf16*)(ws + XB + 3 * WB);
  __bf16* kbuf = (__bf16*)(ws + 2 * XB + 3 * WB);
  __bf16* vt   = (__bf16*)(ws + 3 * XB + 3 * WB);

  // Stage 0: bf16 conversion
  cvt_f32_bf16<<<4096, 256, 0, stream>>>(x,  xb,  NROW * HH);
  cvt_f32_bf16<<<1024, 256, 0, stream>>>(Wq, wqb, HH * HH);
  cvt_f32_bf16<<<1024, 256, 0, stream>>>(Wk, wkb, HH * HH);
  cvt_f32_bf16<<<1024, 256, 0, stream>>>(Wv, wvb, HH * HH);

  // Stage 1: QKV projection + bias + RoPE (v transposed)
  dim3 g1(NROW / 16, HH / 256, 3);
  qkv_rope_kernel<<<g1, 128, 0, stream>>>(xb, wqb, wkb, wvb, bq, bk, bv,
                                          qb, kbuf, vt);

  // Stage 2: streaming attention
  attn_kernel<<<NROW / 16, 256, 0, stream>>>(qb, kbuf, vt, out);
}
